// CovariantChartRouter_27195732918751
// MI455X (gfx1250) — matrix operand order repaired and checked by reading, not compiled, not run on hardware
//
#include <hip/hip_runtime.h>
#include <math.h>

// ---------------- CDNA5 WMMA vector types ----------------
typedef __attribute__((ext_vector_type(16))) __bf16 v16bf;
typedef __attribute__((ext_vector_type(8)))  __bf16 v8bf;
typedef __attribute__((ext_vector_type(8)))  float  v8f;

#define DD   512
#define KK   64
#define ZSTR 520                   // padded bf16 row stride (bank spread, 16B aligned)
#define CEPS 1.001f                // 1 + TRANSPORT_EPS

// LDS layout (bytes)
#define OFF_S   0                  // float S[16][64][65]  (augmented)  266240 B
#define OFF_ZT  266240             // __bf16 zt[16][520]                 16640 B
#define OFF_Q   282880             // float q[16][64]                     4096 B
#define OFF_Y   286976             // float y[16][64]                     4096 B
#define OFF_SC  291072             // float sc[16][64]                    4096 B
#define OFF_R2  295168             // float r2[16]                          64 B
#define OFF_ZU  295232             // float zu[16][8]                      512 B
#define OFF_ZV  295744             // float zv[16][8]                      512 B
#define SMEM_BYTES 296256

// ---- operand builders (contiguous k order k = 16*half + s, 16-bit A/B, wave32) ----
__device__ inline v16bf loadA(const __bf16* __restrict__ zrow, int kbase, int klo_a) {
  v8bf lo = *(const v8bf*)(zrow + kbase + 2 * klo_a);
  v8bf hi = *(const v8bf*)(zrow + kbase + 2 * klo_a + 8);
  return __builtin_shufflevector(lo, hi, 0,1,2,3,4,5,6,7,8,9,10,11,12,13,14,15);
}
__device__ inline v16bf loadB_bf16(const __bf16* __restrict__ wrow, int off) {
  v8bf lo = *(const v8bf*)(wrow + off);
  v8bf hi = *(const v8bf*)(wrow + off + 8);
  return __builtin_shufflevector(lo, hi, 0,1,2,3,4,5,6,7,8,9,10,11,12,13,14,15);
}
__device__ inline v16bf loadB_f32(const float* __restrict__ wrow, int off) {
  const float4* wp = (const float4*)(wrow + off);
  float4 w0 = wp[0], w1 = wp[1], w2 = wp[2], w3 = wp[3];
  v16bf Bv;
  Bv[0]=(__bf16)w0.x;  Bv[1]=(__bf16)w0.y;  Bv[2]=(__bf16)w0.z;  Bv[3]=(__bf16)w0.w;
  Bv[4]=(__bf16)w1.x;  Bv[5]=(__bf16)w1.y;  Bv[6]=(__bf16)w1.z;  Bv[7]=(__bf16)w1.w;
  Bv[8]=(__bf16)w2.x;  Bv[9]=(__bf16)w2.y;  Bv[10]=(__bf16)w2.z; Bv[11]=(__bf16)w2.w;
  Bv[12]=(__bf16)w3.x; Bv[13]=(__bf16)w3.y; Bv[14]=(__bf16)w3.z; Bv[15]=(__bf16)w3.w;
  return Bv;
}
__device__ inline v8f wmma_bf16(v16bf A, v16bf B, v8f C) {
  return __builtin_amdgcn_wmma_f32_16x16x32_bf16(false, A, false, B, (short)0, C,
                                                 false, false);
}

// Single 16x16 tile of z_tile @ W^T over K=512 (16 chunks of 32).
template <bool WSBF>
__device__ inline v8f wmma_rowtile(const __bf16* __restrict__ zt,
                                   const float* __restrict__ Wf,
                                   const __bf16* __restrict__ Wb,
                                   int wrow0, int lane) {
  v8f acc = {};
  const int n     = lane & 15;
  const int klo_a = (lane < 16) ? 0 : 8;
  const int klo_b = (lane < 16) ? 0 : 16;
  const __bf16* zrow = zt + (lane & 15) * ZSTR;
  const float*  wfr  = WSBF ? nullptr : (Wf + (size_t)(wrow0 + n) * DD);
  const __bf16* wbr  = WSBF ? (Wb + (size_t)(wrow0 + n) * DD) : nullptr;
  for (int c = 0; c < 16; ++c) {
    v16bf A = loadA(zrow, c * 32, klo_a);
    v16bf B = WSBF ? loadB_bf16(wbr, c * 32 + klo_b) : loadB_f32(wfr, c * 32 + klo_b);
    acc = wmma_bf16(A, B, acc);
  }
  return acc;
}

// ---- pre-pass: convert w_t and w_z f32 -> bf16 into workspace ----
extern "C" __global__ void convert_weights_kernel(const float* __restrict__ wt,
                                                  const float* __restrict__ wz,
                                                  __bf16* __restrict__ ws,
                                                  long wt_elems, long total) {
  const long i = ((long)blockIdx.x * blockDim.x + threadIdx.x) * 4;
  if (i >= total) return;
  float4 v = (i < wt_elems) ? *(const float4*)(wt + i)
                            : *(const float4*)(wz + (i - wt_elems));
  __bf16* d = ws + i;
  d[0] = (__bf16)v.x; d[1] = (__bf16)v.y; d[2] = (__bf16)v.z; d[3] = (__bf16)v.w;
}

template <bool WSBF>
__global__ void __launch_bounds__(256, 1)
covariant_chart_router_kernel(const float* __restrict__ z,
                              const float* __restrict__ w_z,
                              const float* __restrict__ b_z,
                              const float* __restrict__ g_out,
                              const float* __restrict__ g_u,
                              const float* __restrict__ g_v,
                              const float* __restrict__ cq,
                              const float* __restrict__ w_t,
                              const __bf16* __restrict__ wt_bf,
                              const __bf16* __restrict__ wz_bf,
                              float* __restrict__ out_rw,
                              float* __restrict__ out_kc) {
  extern __shared__ char smem[];
  float*  S    = (float*)(smem + OFF_S);    // 16 matrices, row stride 65 (col 64 = RHS/x)
  __bf16* zt   = (__bf16*)(smem + OFF_ZT);
  float*  qarr = (float*)(smem + OFF_Q);
  float*  ys   = (float*)(smem + OFF_Y);
  float*  sc   = (float*)(smem + OFF_SC);
  float*  r2s  = (float*)(smem + OFF_R2);
  float*  zus  = (float*)(smem + OFF_ZU);
  float*  zvs  = (float*)(smem + OFF_ZV);

  const int tid  = threadIdx.x;
  const int lane = tid & 31;
  const int wave = tid >> 5;
  const int row0 = blockIdx.x * 16;

  // ---- Phase 0: zero small accumulators ----
  if (tid < 16)  r2s[tid] = 0.0f;
  if (tid < 128) { zus[tid] = 0.0f; zvs[tid] = 0.0f; }
  __syncthreads();

  // ---- Phase A: load z tile (f32), store bf16 to LDS, accumulate r2 ----
  {
    const int r  = tid >> 4;
    const int c0 = (tid & 15) * 32;
    const float4* zp = (const float4*)(z + (size_t)(row0 + r) * DD + c0);
    float p = 0.0f;
#pragma unroll
    for (int u = 0; u < 8; ++u) {
      float4 v4 = zp[u];
      p += v4.x*v4.x + v4.y*v4.y + v4.z*v4.z + v4.w*v4.w;
      __bf16* dst = zt + r * ZSTR + c0 + u * 4;
      dst[0] = (__bf16)v4.x; dst[1] = (__bf16)v4.y;
      dst[2] = (__bf16)v4.z; dst[3] = (__bf16)v4.w;
    }
    atomicAdd(&r2s[r], p);
  }
  __syncthreads();

  // ---- Phase A2: zu = z@g_u^T, zv = z@g_v^T  (16 rows x 8 ranks, 2 halves) ----
  {
    const int m = tid >> 4, j = (tid >> 1) & 7, half = tid & 1;
    const float*  gu = g_u + j * DD + half * 256;
    const float*  gv = g_v + j * DD + half * 256;
    const __bf16* zr = zt + m * ZSTR + half * 256;
    float su = 0.0f, sv = 0.0f;
    for (int k = 0; k < 256; ++k) {
      float zz = (float)zr[k];
      su += zz * gu[k];
      sv += zz * gv[k];
    }
    atomicAdd(&zus[m * 8 + j], su);
    atomicAdd(&zvs[m * 8 + j], sv);
  }
  __syncthreads();

  // ---- Phase B: skew GEMM  S16x4096 = z16x512 @ w_t^T  (WMMA bf16) ----
  // 256 N-tiles; each wave owns 32, processed 2-at-a-time sharing the A operand.
  {
    const int n     = lane & 15;
    const int klo_a = (lane < 16) ? 0 : 8;
    const int klo_b = (lane < 16) ? 0 : 16;
    const int mb    = (lane < 16) ? 0 : 8;
    const __bf16* zrow = zt + n * ZSTR;
    for (int u = 0; u < 16; ++u) {
      const int nt0 = wave + 8 * (2 * u);
      const int nt1 = nt0 + 8;
      const float*  wf0 = WSBF ? nullptr : (w_t + (size_t)(nt0 * 16 + n) * DD);
      const float*  wf1 = WSBF ? nullptr : (w_t + (size_t)(nt1 * 16 + n) * DD);
      const __bf16* wb0 = WSBF ? (wt_bf + (size_t)(nt0 * 16 + n) * DD) : nullptr;
      const __bf16* wb1 = WSBF ? (wt_bf + (size_t)(nt1 * 16 + n) * DD) : nullptr;
      v8f acc0 = {}, acc1 = {};
      for (int c = 0; c < 16; ++c) {
        const int kb = c * 32;
        v16bf A  = loadA(zrow, kb, klo_a);
        v16bf B0 = WSBF ? loadB_bf16(wb0, kb + klo_b) : loadB_f32(wf0, kb + klo_b);
        v16bf B1 = WSBF ? loadB_bf16(wb1, kb + klo_b) : loadB_f32(wf1, kb + klo_b);
        acc0 = wmma_bf16(A, B0, acc0);
        acc1 = wmma_bf16(A, B1, acc1);
      }
#pragma unroll
      for (int v = 0; v < 8; ++v) {
        S[(mb + v) * 4160 + (nt0 >> 2) * 65 + (nt0 & 3) * 16 + n] = acc0[v];
        S[(mb + v) * 4160 + (nt1 >> 2) * 65 + (nt1 & 3) * 16 + n] = acc1[v];
      }
    }
  }
  __syncthreads();

  // ---- Phase Q (waves 0-3): q = z@w_z^T + b_z + (zu*zv)@g_out  via WMMA
  // ---- Phase D (waves 4-7): in-place build Bm = c*I - 0.25*(S - S^T)
  if (wave < 4) {
    v8f acc = wmma_rowtile<WSBF>(zt, w_z, wz_bf, wave * 16, lane);
    const int n  = wave * 16 + (lane & 15);
    const int mb = (lane < 16) ? 0 : 8;
    const float bz = b_z[n];
#pragma unroll
    for (int v = 0; v < 8; ++v) {
      const int m = mb + v;
      float g = 0.0f;
#pragma unroll
      for (int j = 0; j < 8; ++j)
        g += zus[m * 8 + j] * zvs[m * 8 + j] * g_out[j * 64 + n];
      const float qv = acc[v] + bz + g;
      qarr[m * 64 + n] = qv;
      S[m * 4160 + n * 65 + 64] = qv;   // augmented RHS column (disjoint from D)
    }
  } else {
    const int t2 = tid - 128;
    for (int idx = t2; idx < 16 * 64 * 64; idx += 128) {
      const int m = idx >> 12, ij = idx & 4095, i = ij >> 6, j = ij & 63;
      float* A = S + m * 4160;
      if (i < j) {
        const float a = A[i * 65 + j], b = A[j * 65 + i];
        const float v = 0.25f * (a - b);
        A[i * 65 + j] = -v;            // Bm = cI - M, M skew
        A[j * 65 + i] =  v;
      } else if (i == j) {
        A[i * 65 + i] = CEPS;
      }
    }
  }
  __syncthreads();

  // ---- Phase E: per-wave Gaussian elimination (no pivoting: sym part = cI > 0),
  //      back-substitution, then y = 2c*x - q.  2 matrices per wave. ----
  for (int mm = 0; mm < 2; ++mm) {
    const int m = wave * 2 + mm;
    float* A = S + m * 4160;
    const int c0 = lane, c1 = lane + 32;
    for (int k = 0; k < 63; ++k) {
      const float inv = 1.0f / A[k * 65 + k];
      const float rk0 = A[k * 65 + c0];
      const float rk1 = A[k * 65 + c1];
      const float rka = A[k * 65 + 64];
      for (int i = k + 1; i < 64; ++i) {
        const float f = A[i * 65 + k] * inv;    // broadcast LDS read
        A[i * 65 + c0] -= f * rk0;              // cols <= k: provably no-op
        A[i * 65 + c1] -= f * rk1;
        if (lane == 0) A[i * 65 + 64] -= f * rka;
      }
    }
    for (int k = 63; k >= 0; --k) {
      float s = 0.0f;
      if (c0 > k) s += A[k * 65 + c0] * A[c0 * 65 + 64];
      if (c1 > k) s += A[k * 65 + c1] * A[c1 * 65 + 64];
      for (int off = 16; off; off >>= 1) s += __shfl_xor(s, off, 32);
      if (lane == 0)
        A[k * 65 + 64] = (A[k * 65 + 64] - s) / A[k * 65 + k];
    }
    const float x0 = A[c0 * 65 + 64], x1 = A[c1 * 65 + 64];
    ys[m * 64 + c0] = 2.0f * CEPS * x0 - qarr[m * 64 + c0];
    ys[m * 64 + c1] = 2.0f * CEPS * x1 - qarr[m * 64 + c1];
  }
  __syncthreads();

  // ---- Phase F: scores[m][n] = cq[n] . y[m]  (16x64 dots of 64) ----
#pragma unroll
  for (int u = 0; u < 4; ++u) {
    const int idx = tid + 256 * u;
    const int m = idx >> 6, n = idx & 63;
    const float* cqr = cq + n * 64;
    const float* yr  = ys + m * 64;
    float s = 0.0f;
    for (int i = 0; i < 64; ++i) s += cqr[i] * yr[i];
    sc[idx] = s;
  }
  __syncthreads();

  // ---- Phase G: temperature, softmax, argmax; 2 rows per wave ----
  for (int mm = 0; mm < 2; ++mm) {
    const int m = wave * 2 + mm;
    const float denom = fmaxf(1.0f - r2s[m], 1e-3f);
    const float itau  = 1.0f / fmaxf(4.0f * denom, 0.01f);  // sqrt(64)/2 = 4
    const int n0 = lane, n1 = lane + 32;
    const float v0 = sc[m * 64 + n0] * itau;
    const float v1 = sc[m * 64 + n1] * itau;
    float mx = fmaxf(v0, v1);
    for (int off = 16; off; off >>= 1) mx = fmaxf(mx, __shfl_xor(mx, off, 32));
    const float e0 = __expf(v0 - mx), e1 = __expf(v1 - mx);
    float sum = e0 + e1;
    for (int off = 16; off; off >>= 1) sum += __shfl_xor(sum, off, 32);
    const float inv = 1.0f / sum;
    const size_t gb = (size_t)(row0 + m) * 64;
    out_rw[gb + n0] = e0 * inv;
    out_rw[gb + n1] = e1 * inv;
    int cand = 64;
    if (v1 == mx) cand = n1;
    if (v0 == mx) cand = n0;                 // first-max tie-break like argmax
    for (int off = 16; off; off >>= 1) {
      int o = __shfl_xor(cand, off, 32);
      cand = (o < cand) ? o : cand;
    }
    if (lane == 0) out_kc[row0 + m] = (float)cand;
  }
}

extern "C" void kernel_launch(void* const* d_in, const int* in_sizes, int n_in,
                              void* d_out, int out_size, void* d_ws, size_t ws_size,
                              hipStream_t stream) {
  const float* z    = (const float*)d_in[0];
  const float* w_z  = (const float*)d_in[1];
  const float* b_z  = (const float*)d_in[2];
  const float* g_o  = (const float*)d_in[3];
  const float* g_u  = (const float*)d_in[4];
  const float* g_v  = (const float*)d_in[5];
  const float* cq   = (const float*)d_in[6];
  const float* w_t  = (const float*)d_in[7];

  const int  Bn = in_sizes[0] / DD;               // 32768
  const long wt_elems = in_sizes[7];              // 4096*512
  const long wz_elems = in_sizes[1];              // 64*512
  const long total    = wt_elems + wz_elems;

  float* out_rw = (float*)d_out;                  // [B, 64] router weights
  float* out_kc = out_rw + (size_t)Bn * 64;       // [B] chart index (as float)

  dim3 grid(Bn / 16), block(256);
  const bool use_bf_ws = ws_size >= (size_t)total * sizeof(__bf16);

  if (use_bf_ws) {
    __bf16* ws_bf = (__bf16*)d_ws;
    const long cthreads = (total + 3) / 4;
    dim3 cgrid((cthreads + 255) / 256);
    convert_weights_kernel<<<cgrid, block, 0, stream>>>(w_t, w_z, ws_bf,
                                                        wt_elems, total);
    covariant_chart_router_kernel<true><<<grid, block, SMEM_BYTES, stream>>>(
        z, w_z, b_z, g_o, g_u, g_v, cq, w_t,
        (const __bf16*)ws_bf, (const __bf16*)(ws_bf + wt_elems), out_rw, out_kc);
  } else {
    covariant_chart_router_kernel<false><<<grid, block, SMEM_BYTES, stream>>>(
        z, w_z, b_z, g_o, g_u, g_v, cq, w_t,
        (const __bf16*)nullptr, (const __bf16*)nullptr, out_rw, out_kc);
  }
}
